// EdgeMLP_51591147160149
// MI455X (gfx1250) — matrix operand order repaired and checked by reading, not compiled, not run on hardware
//
#include <hip/hip_runtime.h>
#include <hip/hip_bf16.h>

typedef _Float16 v16h __attribute__((ext_vector_type(16)));
typedef _Float16 v8h  __attribute__((ext_vector_type(8)));
typedef float    v8f  __attribute__((ext_vector_type(8)));
typedef float    v4f  __attribute__((ext_vector_type(4)));

#define THREADS 256
#define WAVES_PER_BLOCK 8
#define W1T_STRIDE 136   // halves per row: 128 + 8 pad (272B, 16B aligned, conflict-free)
#define W2T_STRIDE 40    // halves per row: 32 + 8 pad  (80B, 16B aligned, conflict-free)
#define HT_STRIDE  40    // halves per row of the per-wave h1 tile

__global__ void __launch_bounds__(THREADS) edge_mlp_zero(float* out, int n) {
    int i = blockIdx.x * blockDim.x + threadIdx.x;
    if (i < n) out[i] = 0.0f;
}

// Build one K=32 chunk of the 16-bit A operand from a node's feature row.
// A layout (16-bit A 16x32): lanes 0-15 hold M=lane, VGPR0..3: K=0..7, VGPR4..7: K=16..23;
// lanes 16-31 hold M=lane-16, VGPR0..3: K=8..15, VGPR4..7: K=24..31.  => per lane-half h:
// halves [0..7]  <- feat[8h .. 8h+7],  halves [8..15] <- feat[16+8h .. 16+8h+7].
__device__ __forceinline__ v16h make_a_from_x(const float* __restrict__ p, int h) {
    const float* q0 = p + 8 * h;
    const float* q1 = p + 16 + 8 * h;
    v4f f0 = *(const v4f*)(q0);
    v4f f1 = *(const v4f*)(q0 + 4);
    v4f f2 = *(const v4f*)(q1);
    v4f f3 = *(const v4f*)(q1 + 4);
    v16h a;
    a[0]=(_Float16)f0[0]; a[1]=(_Float16)f0[1]; a[2]=(_Float16)f0[2]; a[3]=(_Float16)f0[3];
    a[4]=(_Float16)f1[0]; a[5]=(_Float16)f1[1]; a[6]=(_Float16)f1[2]; a[7]=(_Float16)f1[3];
    a[8]=(_Float16)f2[0]; a[9]=(_Float16)f2[1]; a[10]=(_Float16)f2[2]; a[11]=(_Float16)f2[3];
    a[12]=(_Float16)f3[0]; a[13]=(_Float16)f3[1]; a[14]=(_Float16)f3[2]; a[15]=(_Float16)f3[3];
    return a;
}

// halves [base..base+7] and [base+16..base+23] from LDS (two ds_load_b128) -> v16h
__device__ __forceinline__ v16h lds16(const _Float16* base) {
    v8h lo = *(const v8h*)(base);
    v8h hi = *(const v8h*)(base + 16);
    v16h b;
#pragma unroll
    for (int i = 0; i < 8; ++i) { b[i] = lo[i]; b[8 + i] = hi[i]; }
    return b;
}

__global__ void __launch_bounds__(THREADS) edge_mlp_spmm(
    const float* __restrict__ x, const long long* __restrict__ ei,
    const float* __restrict__ u,
    const float* __restrict__ W1, const float* __restrict__ b1,
    const float* __restrict__ W2, const float* __restrict__ b2,
    const float* __restrict__ W3, const float* __restrict__ b3,
    float* __restrict__ out, int E, int N)
{
    __shared__ _Float16 s_w1t[32 * W1T_STRIDE];      // W1^T as f16, padded rows
    __shared__ _Float16 s_w2t[32 * W2T_STRIDE];      // W2^T as f16, padded rows
    __shared__ float s_b1[32], s_b2[32], s_w3[32], s_b3;
    __shared__ _Float16 s_h[WAVES_PER_BLOCK * 16 * HT_STRIDE]; // per-wave h1 tile (16x32 f16)
    __shared__ float s_wv[WAVES_PER_BLOCK * 16];               // per-wave edge weights

    const int tid = threadIdx.x;

    // ---- stage weights (transposed, f16) into LDS once per block ----
    for (int i = tid; i < 128 * 32; i += THREADS) {
        int k = i >> 5, n = i & 31;                  // W1[k][n], row-major 128x32
        s_w1t[n * W1T_STRIDE + k] = (_Float16)W1[i];
    }
    for (int i = tid; i < 32 * 32; i += THREADS) {
        int k = i >> 5, n = i & 31;                  // W2[k][n], row-major 32x32
        s_w2t[n * W2T_STRIDE + k] = (_Float16)W2[i];
    }
    if (tid < 32) { s_b1[tid] = b1[tid]; s_b2[tid] = b2[tid]; s_w3[tid] = W3[tid]; }
    if (tid == 0) s_b3 = b3[0];
    __syncthreads();

    const int lane  = tid & 31;
    const int ln16  = lane & 15;
    const int h     = lane >> 4;
    const int wslot = tid >> 5;
    const int waveG = (blockIdx.x * blockDim.x + tid) >> 5;
    const int totalWaves = (gridDim.x * blockDim.x) >> 5;

    // ---- persistent B operands ----
    // B layout (16-bit B KxN): lanes 0-15: N=lane,  K=0..7,16..23 across VGPR halves;
    //                          lanes 16-31: N=lane-16, K=8..15,24..31. Per K=32 chunk c:
    // halves of (W^T row N) at 32c + 8h + [0..7] and 32c + 16 + 8h + [0..7].
    v16h B1[2][4];
#pragma unroll
    for (int n0 = 0; n0 < 2; ++n0)
#pragma unroll
        for (int c = 0; c < 4; ++c)
            B1[n0][c] = lds16(&s_w1t[(n0 * 16 + ln16) * W1T_STRIDE + 32 * c + 8 * h]);
    v16h B2[2];
#pragma unroll
    for (int n0 = 0; n0 < 2; ++n0)
        B2[n0] = lds16(&s_w2t[(n0 * 16 + ln16) * W2T_STRIDE + 8 * h]);

    const float bias1_0 = s_b1[ln16], bias1_1 = s_b1[16 + ln16];
    const float bias2_0 = s_b2[ln16], bias2_1 = s_b2[16 + ln16];
    const float w3_0 = s_w3[ln16],   w3_1 = s_w3[16 + ln16];
    const float b3v = s_b3;

    _Float16* hrow = &s_h[wslot * 16 * HT_STRIDE];
    float*    wv   = &s_wv[wslot * 16];

    const int numTiles = (E + 15) >> 4;
    for (int t = waveG; t < numTiles; t += totalWaves) {
        int e = t * 16 + ln16;
        bool valid = e < E;
        long long eC = valid ? (long long)e : 0;
        int row = (int)ei[eC];
        int col = (int)ei[(long long)E + eC];

        // ---- layer 1: [16x128] x [128x32] via 8 WMMAs ----
        const float* prow = x + (long long)row * 64;
        const float* pcol = x + (long long)col * 64;
        v16h A0 = make_a_from_x(prow,      h);  // K  0..31  (x[row] 0..31)
        v16h A1 = make_a_from_x(prow + 32, h);  // K 32..63  (x[row] 32..63)
        v16h A2c = make_a_from_x(pcol,      h); // K 64..95  (x[col] 0..31)
        v16h A3 = make_a_from_x(pcol + 32, h);  // K 96..127 (x[col] 32..63)

        v8f acc0, acc1;
#pragma unroll
        for (int r = 0; r < 8; ++r) { acc0[r] = bias1_0; acc1[r] = bias1_1; }
        acc0 = __builtin_amdgcn_wmma_f32_16x16x32_f16(false, A0,  false, B1[0][0], (short)0, acc0, false, false);
        acc0 = __builtin_amdgcn_wmma_f32_16x16x32_f16(false, A1,  false, B1[0][1], (short)0, acc0, false, false);
        acc0 = __builtin_amdgcn_wmma_f32_16x16x32_f16(false, A2c, false, B1[0][2], (short)0, acc0, false, false);
        acc0 = __builtin_amdgcn_wmma_f32_16x16x32_f16(false, A3,  false, B1[0][3], (short)0, acc0, false, false);
        acc1 = __builtin_amdgcn_wmma_f32_16x16x32_f16(false, A0,  false, B1[1][0], (short)0, acc1, false, false);
        acc1 = __builtin_amdgcn_wmma_f32_16x16x32_f16(false, A1,  false, B1[1][1], (short)0, acc1, false, false);
        acc1 = __builtin_amdgcn_wmma_f32_16x16x32_f16(false, A2c, false, B1[1][2], (short)0, acc1, false, false);
        acc1 = __builtin_amdgcn_wmma_f32_16x16x32_f16(false, A3,  false, B1[1][3], (short)0, acc1, false, false);

        // ---- relu + reshape C-layout -> A-layout through per-wave LDS tile ----
        // C layout: VGPR r, lane: element (M = r + 8h, N = ln16 [+16 for acc1])
#pragma unroll
        for (int r = 0; r < 8; ++r) {
            int M = r + 8 * h;
            float v0 = acc0[r] > 0.f ? acc0[r] : 0.f;
            float v1 = acc1[r] > 0.f ? acc1[r] : 0.f;
            hrow[M * HT_STRIDE + ln16]      = (_Float16)v0;
            hrow[M * HT_STRIDE + 16 + ln16] = (_Float16)v1;
        }
        asm volatile("s_wait_dscnt 0x0" ::: "memory");

        // layer-2 A operand (K=32 chunk) from h tile, A layout: row M = ln16
        v8h lo = *(const v8h*)&hrow[ln16 * HT_STRIDE + 8 * h];
        v8h hi = *(const v8h*)&hrow[ln16 * HT_STRIDE + 16 + 8 * h];
        v16h Ah;
#pragma unroll
        for (int i = 0; i < 8; ++i) { Ah[i] = lo[i]; Ah[8 + i] = hi[i]; }

        // ---- layer 2: [16x32] x [32x32] via 2 WMMAs ----
        v8f acc20, acc21;
#pragma unroll
        for (int r = 0; r < 8; ++r) { acc20[r] = bias2_0; acc21[r] = bias2_1; }
        acc20 = __builtin_amdgcn_wmma_f32_16x16x32_f16(false, Ah, false, B2[0], (short)0, acc20, false, false);
        acc21 = __builtin_amdgcn_wmma_f32_16x16x32_f16(false, Ah, false, B2[1], (short)0, acc21, false, false);

        // ---- layer 3: relu, dot with W3, butterfly reduce over each 16-lane half ----
        float tr[8];
#pragma unroll
        for (int r = 0; r < 8; ++r) {
            float v0 = acc20[r] > 0.f ? acc20[r] : 0.f;
            float v1 = acc21[r] > 0.f ? acc21[r] : 0.f;
            tr[r] = v0 * w3_0 + v1 * w3_1;
        }
#pragma unroll
        for (int m = 1; m < 16; m <<= 1) {
#pragma unroll
            for (int r = 0; r < 8; ++r) tr[r] += __shfl_xor(tr[r], m, 32);
        }
        // lanes (ln16==0) of each half hold edge weights for M = 8h + r
        if (ln16 == 0) {
#pragma unroll
            for (int r = 0; r < 8; ++r) wv[8 * h + r] = tr[r] + b3v;
        }
        asm volatile("s_wait_dscnt 0x0" ::: "memory");

        // ---- scatter: f[row[e]] += w_e * u[col[e]] ----
        if (lane < 16 && valid) {
            float w = wv[lane];               // edge = t*16 + lane, matches this lane's row/col
            atomicAdd(&out[row], w * u[col]);
        }
    }
}

extern "C" void kernel_launch(void* const* d_in, const int* in_sizes, int n_in,
                              void* d_out, int out_size, void* d_ws, size_t ws_size,
                              hipStream_t stream) {
    const float*     x  = (const float*)d_in[0];
    const long long* ei = (const long long*)d_in[1];
    const float*     u  = (const float*)d_in[2];
    const float*     W1 = (const float*)d_in[3];
    const float*     b1 = (const float*)d_in[4];
    const float*     W2 = (const float*)d_in[5];
    const float*     b2 = (const float*)d_in[6];
    const float*     W3 = (const float*)d_in[7];
    const float*     b3 = (const float*)d_in[8];
    float* out = (float*)d_out;

    int E = in_sizes[1] / 2;
    int N = out_size;

    edge_mlp_zero<<<(N + THREADS - 1) / THREADS, THREADS, 0, stream>>>(out, N);

    int numTiles = (E + 15) / 16;
    int blocks = (numTiles + WAVES_PER_BLOCK - 1) / WAVES_PER_BLOCK;
    if (blocks > 768) blocks = 768;   // grid-stride over edge tiles
    if (blocks < 1) blocks = 1;
    edge_mlp_spmm<<<blocks, THREADS, 0, stream>>>(x, ei, u, W1, b1, W2, b2, W3, b3, out, E, N);
}
